// DualBranchDecoder_86509231276770
// MI455X (gfx1250) — matrix operand order repaired and checked by reading, not compiled, not run on hardware
//
#include <hip/hip_runtime.h>
#include <hip/hip_bf16.h>
#include <math.h>

#define CC 128
#define HH 512
#define TT 512
#define BB 8
#define KK 31
#define FF 257
#define EPSV 1e-5f

typedef __attribute__((ext_vector_type(16))) __bf16 v16bf;
typedef __attribute__((ext_vector_type(8)))  float  v8f;

__constant__ int BAND_W_c[KK] = {2,3,3,3,3,3,3,3,3,3,3,
                                 8,8,8,8,8,8,8,8,8,8,8,8,
                                 16,16,16,16,16,16,16, 17};
__constant__ int BAND_OFF_c[KK] = {0,2,5,8,11,14,17,20,23,26,29,
                                   32,40,48,56,64,72,80,88,96,104,112,120,
                                   128,144,160,176,192,208,224, 240};

__device__ __forceinline__ unsigned short f2bf(float f) {
  unsigned int u = __float_as_uint(f);
  unsigned int r = u + 0x7FFFu + ((u >> 16) & 1u);   // round-to-nearest-even
  return (unsigned short)(r >> 16);
}

// Branch-free tanh on the native v_exp_f32 / v_rcp_f32 units.
// tanh(x) = sign(x) * (1 - 2/(exp(2|x|)+1)); large |x| -> exp=inf -> rcp=0 -> 1.
__device__ __forceinline__ float fast_tanh(float x) {
  float ax = __builtin_fabsf(x);
  float e  = __builtin_amdgcn_exp2f(ax * 2.885390081777927f);   // exp(2|x|)
  float t  = 1.f - 2.f * __builtin_amdgcn_rcpf(e + 1.f);
  return __builtin_copysignf(t, x);
}

// sigmoid(x) = 1/(1+exp(-x)), branch-free
__device__ __forceinline__ float fast_sigmoid(float x) {
  float e = __builtin_amdgcn_exp2f(-1.4426950408889634f * x);
  return __builtin_amdgcn_rcpf(1.f + e);
}

__device__ __forceinline__ v8f wmma_bf16(v16bf a, v16bf b, v8f c) {
  return __builtin_amdgcn_wmma_f32_16x16x32_bf16(false, a, false, b, (short)0, c, false, false);
}

// ---------------------------------------------------------------- stats ----
__global__ void zero_stats_kernel(float* ssum, float* ssq) {
  int i = blockIdx.x * blockDim.x + threadIdx.x;
  if (i < 2 * BB * KK) { ssum[i] = 0.f; ssq[i] = 0.f; }
}

// grid = 2*B*C blocks; each reduces a contiguous [T,K] slab for one (br,b,c)
__global__ void stats_partial_kernel(const float* __restrict__ magf,
                                     const float* __restrict__ phf,
                                     float* __restrict__ ssum,
                                     float* __restrict__ ssq) {
  int blk = blockIdx.x;
  int br  = blk / (BB * CC);
  int rem = blk % (BB * CC);
  int b = rem / CC, c = rem % CC;
  const float* feat = br ? phf : magf;
  const float* slab = feat + ((size_t)(b * CC + c) * TT) * KK;

  float s[KK], q[KK];
#pragma unroll
  for (int k = 0; k < KK; k++) { s[k] = 0.f; q[k] = 0.f; }
  for (int t = threadIdx.x; t < TT; t += blockDim.x) {
    const float* row = slab + (size_t)t * KK;
#pragma unroll
    for (int k = 0; k < KK; k++) { float v = row[k]; s[k] += v; q[k] += v * v; }
  }

  __shared__ float acc[2 * KK];
  for (int i = threadIdx.x; i < 2 * KK; i += blockDim.x) acc[i] = 0.f;
  __syncthreads();

  int lane = threadIdx.x & 31;
#pragma unroll
  for (int k = 0; k < KK; k++) {
    float vs = s[k], vq = q[k];
    for (int off = 16; off > 0; off >>= 1) {
      vs += __shfl_down(vs, off, 32);
      vq += __shfl_down(vq, off, 32);
    }
    if (lane == 0) { atomicAdd(&acc[k], vs); atomicAdd(&acc[KK + k], vq); }
  }
  __syncthreads();
  for (int i = threadIdx.x; i < KK; i += blockDim.x) {
    atomicAdd(&ssum[(br * BB + b) * KK + i], acc[i]);
    atomicAdd(&ssq [(br * BB + b) * KK + i], acc[KK + i]);
  }
}

__global__ void stats_finalize_kernel(const float* ssum, const float* ssq,
                                      float* mean, float* rs) {
  int i = blockIdx.x * blockDim.x + threadIdx.x;
  if (i < 2 * BB * KK) {
    const float N = (float)(CC * TT);
    float m = ssum[i] / N;
    float v = ssq[i] / N - m * m;
    mean[i] = m;
    rs[i] = rsqrtf(v + EPSV);
  }
}

// ------------------------------------------ A-matrix fragment builder ------
// Output layout: afrag[br][k][mgroup(256)][ks(4)][lane(32)][16 halves]
// (exact v_wmma 16-bit A-matrix 16x32 VGPR layout, one 32B load per lane)
__global__ void build_afrag_kernel(const float* __restrict__ magf,
                                   const float* __restrict__ phf,
                                   const float* __restrict__ mgamma,
                                   const float* __restrict__ mbeta,
                                   const float* __restrict__ pgamma,
                                   const float* __restrict__ pbeta,
                                   const float* __restrict__ mean,
                                   const float* __restrict__ rs,
                                   unsigned short* __restrict__ afrag) {
  extern __shared__ unsigned short lds[];   // [KK][2048]
  int blk = blockIdx.x;
  int br  = blk / (BB * 32);
  int rem = blk % (BB * 32);
  int b = rem / 32, g = rem % 32;           // g: group of 16 t values
  const float* feat  = br ? phf : magf;
  const float* gamma = br ? pgamma : mgamma;
  const float* beta  = br ? pbeta : mbeta;

  __shared__ float smean[KK], srs[KK];
  if (threadIdx.x < KK) {
    smean[threadIdx.x] = mean[(br * BB + b) * KK + threadIdx.x];
    srs[threadIdx.x]   = rs[(br * BB + b) * KK + threadIdx.x];
  }
  __syncthreads();

  const int t0 = g * 16;
  for (int i = threadIdx.x; i < CC * 16 * KK; i += blockDim.x) {
    int c = i / (16 * KK);
    int r = i % (16 * KK);
    int tl = r / KK;                        // 0..15
    int k  = r % KK;
    float x  = feat[((size_t)(b * CC + c) * TT + (t0 + tl)) * KK + k];
    float xn = (x - smean[k]) * srs[k] * gamma[k * CC + c] + beta[k * CC + c];
    int ks = c >> 5, remc = c & 31;
    int laneH, j;
    if      (remc <  8) { laneH = 0; j = remc; }
    else if (remc < 16) { laneH = 1; j = remc - 8; }
    else if (remc < 24) { laneH = 0; j = remc - 8; }
    else                { laneH = 1; j = remc - 16; }
    int lane = laneH * 16 + tl;
    lds[k * 2048 + ks * 512 + lane * 16 + j] = f2bf(xn);
  }
  __syncthreads();

  int gg = b * 32 + g;
  for (int i = threadIdx.x; i < KK * 256; i += blockDim.x) {
    int k = i >> 8, off = i & 255;
    uint4* dst = (uint4*)(afrag + ((size_t)(br * KK + k) * 256 + gg) * 2048);
    dst[off] = ((const uint4*)(lds + k * 2048))[off];
  }
}

// ------------------------------------------ W1 fragment builder ------------
// w1frag[br][k][nt(32)][ks(4)][lane(32)][16 halves]  (B-matrix 32x16 layout)
__global__ void build_w1frag_kernel(const float* __restrict__ mW1,
                                    const float* __restrict__ pW1,
                                    unsigned short* __restrict__ w1frag) {
  int i = blockIdx.x * blockDim.x + threadIdx.x;
  if (i >= 2 * KK * 32 * 4 * 32) return;
  int lane = i & 31;
  int ks   = (i >> 5) & 3;
  int nt   = (i >> 7) & 31;
  int bk   = i >> 12;                       // br*KK + k
  int k = bk % KK, br = bk / KK;
  const float* W1 = br ? pW1 : mW1;
  int h = nt * 16 + (lane & 15);
  int cbase = ks * 32 + ((lane < 16) ? 0 : 8);
  unsigned short* out = w1frag + (size_t)i * 16;
  const float* src = W1 + ((size_t)k * HH + h) * CC + cbase;
#pragma unroll
  for (int j = 0; j < 16; j++) out[j] = f2bf(src[j < 8 ? j : j + 8]);
}

// ------------------------------------------ W2 fragment builder ------------
// w2frag[br][tile(32)][ks2(16)][lane(32)][16 halves], zero-padded columns
__global__ void build_w2frag_kernel(const float* __restrict__ mW2,
                                    const float* __restrict__ pW2,
                                    const float* __restrict__ mb2,
                                    const float* __restrict__ pb2,
                                    unsigned short* __restrict__ w2frag,
                                    float* __restrict__ b2t) {
  int i = blockIdx.x * blockDim.x + threadIdx.x;
  if (i >= 2 * 32 * 16 * 32) return;
  int lane = i & 31;
  int ks2  = (i >> 5) & 15;
  int tile = (i >> 9) & 31;
  int br   = i >> 14;
  const float* W2 = br ? pW2 : mW2;
  const float* b2 = br ? pb2 : mb2;
  int kb = (tile >= 30) ? 30 : tile;
  int jt = (tile >= 30) ? (tile - 30) : 0;
  int n  = lane & 15;
  int fl = jt * 16 + n;
  int width = BAND_W_c[kb];
  bool valid = fl < width;
  int f = BAND_OFF_c[kb] + (valid ? fl : 0);
  int h0 = ks2 * 32 + ((lane < 16) ? 0 : 8);
  unsigned short* out = w2frag + (size_t)i * 16;
  const float* src = W2 + (size_t)f * HH + h0;
#pragma unroll
  for (int j = 0; j < 16; j++) out[j] = valid ? f2bf(src[j < 8 ? j : j + 8]) : (unsigned short)0;
  if (ks2 == 0 && lane < 16) b2t[(br * 32 + tile) * 16 + n] = valid ? b2[f] : 0.f;
}

// ------------------------------------------ fused fc1 (+tanh) + fc2 -------
// grid = 2*31*32 blocks of 256 threads (8 waves). 130 KB dynamic LDS holds
// the full 128x512 h1 tile (bf16, padded rows) so h1 never hits HBM.
__global__ void __launch_bounds__(256, 1)
fc_kernel(const unsigned short* __restrict__ afrag,
          const unsigned short* __restrict__ w1frag,
          const unsigned short* __restrict__ w2frag,
          const float* __restrict__ b2t,
          const float* __restrict__ mb1,
          const float* __restrict__ pb1,
          float* __restrict__ logits) {
  extern __shared__ unsigned short Hs[];    // [128][520] bf16 (pad kills bank conflicts)
  const int HSTRIDE = 520;
  int blk = blockIdx.x;
  int br  = blk / (KK * 32);
  int rem = blk % (KK * 32);
  int k = rem / 32, mt = rem % 32;
  int tid  = threadIdx.x;
  int wave = tid >> 5, lane = tid & 31;
  int laneHalf = lane & 15, hi = lane >> 4;
  const float* b1 = br ? pb1 : mb1;

  union V16 { v16bf v; uint4 u[2]; };

  // Preload this wave's 16 rows of A for all 4 k-steps (one 32B load each)
  const v16bf* Ab = (const v16bf*)(afrag +
      ((size_t)(br * KK + k) * 256 + (mt * 8 + wave)) * 2048);
  v16bf a0 = Ab[0 * 32 + lane];
  v16bf a1 = Ab[1 * 32 + lane];
  v16bf a2 = Ab[2 * 32 + lane];
  v16bf a3 = Ab[3 * 32 + lane];
  const v16bf* Wb = (const v16bf*)(w1frag + (size_t)(br * KK + k) * 65536);

  // Prefetch the W2 fragments this block will need in phase 2.
  {
    const unsigned short* W2p = w2frag + (size_t)(br * 32 + k) * 8192;
    __builtin_prefetch(W2p, 0, 0);
    __builtin_prefetch(W2p + 4096, 0, 0);
  }

  int rowbase = wave * 16 + hi * 8;

  // ---- fc1: 128x512 = 32 n-tiles x 4 k-steps of v_wmma_f32_16x16x32_bf16
  for (int nt = 0; nt < 32; ++nt) {
    v8f acc = {0.f, 0.f, 0.f, 0.f, 0.f, 0.f, 0.f, 0.f};
    acc = wmma_bf16(a0, Wb[(nt * 4 + 0) * 32 + lane], acc);
    acc = wmma_bf16(a1, Wb[(nt * 4 + 1) * 32 + lane], acc);
    acc = wmma_bf16(a2, Wb[(nt * 4 + 2) * 32 + lane], acc);
    acc = wmma_bf16(a3, Wb[(nt * 4 + 3) * 32 + lane], acc);
    int h = nt * 16 + laneHalf;
    float bias = b1[k * HH + h];
#pragma unroll
    for (int r = 0; r < 8; r++) {
      float v = fast_tanh(acc[r] + bias);
      Hs[(rowbase + r) * HSTRIDE + h] = f2bf(v);
    }
  }
  __syncthreads();

  // ---- fc2: h1 (LDS) x W2 fragments (global, L2-resident)
  int width = BAND_W_c[k];
  int foff  = BAND_OFF_c[k];
  int ntl   = (k == 30) ? 2 : 1;
  const unsigned short* hrow = Hs + (size_t)(wave * 16 + laneHalf) * HSTRIDE;
  for (int jt = 0; jt < ntl; ++jt) {
    int tile = k + jt;
    const unsigned short* W2b = w2frag + (size_t)(br * 32 + tile) * 8192;
    v8f acc = {0.f, 0.f, 0.f, 0.f, 0.f, 0.f, 0.f, 0.f};
    for (int ks2 = 0; ks2 < 16; ++ks2) {
      V16 av;
      int h0 = ks2 * 32 + (hi ? 8 : 0);
      av.u[0] = *(const uint4*)(hrow + h0);
      av.u[1] = *(const uint4*)(hrow + h0 + 16);
      const v16bf* bf = (const v16bf*)(W2b + (ks2 * 32 + lane) * 16);
      acc = wmma_bf16(av.v, *bf, acc);
    }
    int n  = laneHalf;
    int fl = jt * 16 + n;
    if (fl < width) {
      float bias = b2t[(br * 32 + tile) * 16 + n];
      int f = foff + fl;
#pragma unroll
      for (int r = 0; r < 8; r++) {
        int m = mt * 128 + rowbase + r;
        int b = m >> 9, t = m & 511;
        logits[(size_t)((br * BB + b) * TT + t) * FF + f] = acc[r] + bias;
      }
    }
  }
}

// ------------------------------------------ final pointwise ---------------
__global__ void final_kernel(const float* __restrict__ logits,
                             const float* __restrict__ noisy_mag,
                             const float* __restrict__ noisy_phase,
                             float* __restrict__ out) {
  const float PI = 3.14159265358979323846f;
  const float TWO_PI = 6.28318530717958647692f;
  size_t i = (size_t)blockIdx.x * blockDim.x + threadIdx.x;
  size_t total = (size_t)BB * FF * TT;
  if (i >= total) return;
  int t = (int)(i % TT);
  int f = (int)((i / TT) % FF);
  int b = (int)(i / ((size_t)TT * FF));
  size_t lidx = (size_t)(b * TT + t) * FF + f;
  float ml = logits[lidx];
  float pl = logits[(size_t)BB * TT * FF + lidx];
  float mask = fast_sigmoid(ml);
  float po = fast_tanh(pl) * PI;
  float em = noisy_mag[i] * mask;
  float ep = noisy_phase[i] + po + PI;
  ep = ep - floorf(ep * (1.f / TWO_PI)) * TWO_PI;   // remainder -> [0, 2pi)
  ep = ep - PI;
  float s, c;
  __sincosf(ep, &s, &c);
  out[2 * i]     = em * c;
  out[2 * i + 1] = em * s;
}

// ------------------------------------------ host launcher -----------------
extern "C" void kernel_launch(void* const* d_in, const int* in_sizes, int n_in,
                              void* d_out, int out_size, void* d_ws, size_t ws_size,
                              hipStream_t stream) {
  const float* magf   = (const float*)d_in[0];
  const float* phf    = (const float*)d_in[1];
  const float* nmag   = (const float*)d_in[2];
  const float* nph    = (const float*)d_in[3];
  const float* mgamma = (const float*)d_in[4];
  const float* mbeta  = (const float*)d_in[5];
  const float* mW1    = (const float*)d_in[6];
  const float* mb1    = (const float*)d_in[7];
  const float* mW2    = (const float*)d_in[8];
  const float* mb2    = (const float*)d_in[9];
  const float* pgamma = (const float*)d_in[10];
  const float* pbeta  = (const float*)d_in[11];
  const float* pW1    = (const float*)d_in[12];
  const float* pb1    = (const float*)d_in[13];
  const float* pW2    = (const float*)d_in[14];
  const float* pb2    = (const float*)d_in[15];

  char* ws = (char*)d_ws;
  float* ssum = (float*)(ws + 0);
  float* ssq  = (float*)(ws + 2048);
  float* mean = (float*)(ws + 4096);
  float* rs   = (float*)(ws + 6144);
  unsigned short* afrag = (unsigned short*)(ws + 8192);
  size_t off = 8192 + (size_t)2 * KK * 256 * 2048 * 2;       // +65.0 MB
  unsigned short* w1frag = (unsigned short*)(ws + off);
  off += (size_t)2 * KK * 65536 * 2;                          // +8.1 MB
  unsigned short* w2frag = (unsigned short*)(ws + off);
  off += (size_t)2 * 32 * 8192 * 2;                           // +1.0 MB
  float* b2t = (float*)(ws + off);
  off += (size_t)2 * 32 * 16 * 4;
  float* logits = (float*)(ws + off);                         // +8.4 MB

  zero_stats_kernel<<<2, 256, 0, stream>>>(ssum, ssq);
  stats_partial_kernel<<<2 * BB * CC, 256, 0, stream>>>(magf, phf, ssum, ssq);
  stats_finalize_kernel<<<2, 256, 0, stream>>>(ssum, ssq, mean, rs);

  size_t alds = (size_t)KK * 2048 * 2;                        // 124 KB
  hipFuncSetAttribute(reinterpret_cast<const void*>(build_afrag_kernel),
                      hipFuncAttributeMaxDynamicSharedMemorySize, (int)alds);
  build_afrag_kernel<<<2 * BB * 32, 256, alds, stream>>>(
      magf, phf, mgamma, mbeta, pgamma, pbeta, mean, rs, afrag);

  build_w1frag_kernel<<<(2 * KK * 32 * 4 * 32 + 255) / 256, 256, 0, stream>>>(
      mW1, pW1, w1frag);
  build_w2frag_kernel<<<(2 * 32 * 16 * 32 + 255) / 256, 256, 0, stream>>>(
      mW2, pW2, mb2, pb2, w2frag, b2t);

  size_t fclds = (size_t)128 * 520 * 2;                       // 130 KB
  hipFuncSetAttribute(reinterpret_cast<const void*>(fc_kernel),
                      hipFuncAttributeMaxDynamicSharedMemorySize, (int)fclds);
  fc_kernel<<<2 * KK * 32, 256, fclds, stream>>>(
      afrag, w1frag, w2frag, b2t, mb1, pb1, logits);

  size_t total = (size_t)BB * FF * TT;
  final_kernel<<<(int)((total + 255) / 256), 256, 0, stream>>>(
      logits, nmag, nph, (float*)d_out);
}
